// ConvGRU_36859409334341
// MI455X (gfx1250) — matrix operand order repaired and checked by reading, not compiled, not run on hardware
//
#include <hip/hip_runtime.h>
#include <hip/hip_bf16.h>

// ---------------- problem constants ----------------
constexpr int T_  = 10;
constexpr int NB  = 4;
constexpr int CIN = 64;
constexpr int C   = 128;     // COUT
constexpr int C2  = 256;     // 2*C
constexpr int C3  = 384;     // 3*C
constexpr int H_  = 64;
constexpr int W_  = 64;
constexpr int HW  = 64 * 64;
constexpr float BN_EPS = 1e-5f;

typedef __attribute__((ext_vector_type(16))) __bf16 v16bf;
typedef __attribute__((ext_vector_type(8)))  __bf16 v8bf;
typedef __attribute__((ext_vector_type(8)))  float  v8f;
typedef __attribute__((ext_vector_type(4)))  float  v4f;

__device__ __forceinline__ __bf16 f2bf(float f) {
    unsigned int u = __builtin_bit_cast(unsigned int, f);
    unsigned int r = (u + 0x7FFFu + ((u >> 16) & 1u)) >> 16;
    unsigned short s = (unsigned short)r;
    return __builtin_bit_cast(__bf16, s);
}
__device__ __forceinline__ float sigmoidf_(float x) { return 1.0f / (1.0f + __expf(-x)); }
__device__ __forceinline__ int clampi(int v, int lo, int hi) {
    return v < lo ? lo : (v > hi ? hi : v);
}

// ---------------- layout conversion kernels ----------------
// x: NCHW f32 -> NHWC bf16 : xT[tn][pix][cin]
__global__ void cvt_x_nhwc(const float* __restrict__ src, __bf16* __restrict__ dst, int n) {
    int i = blockIdx.x * 256 + threadIdx.x;
    if (i >= n) return;
    int pix = i & (HW - 1);
    int r   = i >> 12;
    int ci  = r & (CIN - 1);
    int tn  = r >> 6;
    dst[((size_t)tn * HW + pix) * CIN + ci] = f2bf(src[i]);
}

// w[cout][cin][kh][kw] f32 -> wbf[kh*ksz+kw][cout][cin] bf16  (k-major rows)
__global__ void cvt_w_kmaj(const float* __restrict__ w, __bf16* __restrict__ wbf,
                           int cout, int cin, int ksz) {
    int i = blockIdx.x * 256 + threadIdx.x;
    int total = cout * cin * ksz * ksz;
    if (i >= total) return;
    int kw = i % ksz; int t0 = i / ksz;
    int kh = t0 % ksz; int t1 = t0 / ksz;
    int ci = t1 % cin; int co = t1 / cin;
    wbf[((size_t)(kh * ksz + kw) * cout + co) * cin + ci] = f2bf(w[i]);
}

// ---------------- WMMA implicit-GEMM tile engine ----------------
// 256 threads = 8 waves. Block tile = 128 cout x 64 pixels (one image row).
// Wave (wm 0..3 over M, wn 0..1 over N) owns 32x32 -> four 16x16 accumulators.
// Double-buffered LDS (24KB) via self-toggling pointers (no unroll, no acc
// copies), one barrier per K-chunk, b128 staging + b128 fragment loads,
// strength-reduced global addresses, branchless/predicated im2col.
template <int CINL, int KSZ, int PAD>
__device__ __forceinline__ void conv_tile_acc(
        const __bf16* __restrict__ wbf, int COUTL,
        const __bf16* __restrict__ xT,           // [HW][CINL] for this image
        int m0, int py0, v8f (&acc)[2][2]) {
    __shared__ __bf16 Asub[2][128][32];   // [buf][m][k]  8KB x2
    __shared__ __bf16 Bsub[2][64][32];    // [buf][n][k]  4KB x2

    const int tid  = threadIdx.x;
    const int lane = tid & 31;
    const int wave = tid >> 5;
    const int wm   = wave & 3;
    const int wn   = wave >> 2;
    const int half = lane >> 4;
    const int lm   = lane & 15;

    constexpr int NCH = KSZ * KSZ * (CINL / 32);
    // A-offset deltas (elements), compile-time constants
    constexpr size_t A_STEP = 32;
    const size_t A_WRAP = (size_t)COUTL * CINL - (CINL - 32);

    // staging roles: 8 bf16 per thread per tile-part
    const int rB  = (tid * 8) >> 5;      // 0..63
    const int kB  = (tid * 8) & 31;      // 0,8,16,24

    // hoisted per-thread global base pointers
    const __bf16* aptr0 = wbf + (size_t)(m0 + rB)      * CINL + kB;
    const __bf16* aptr1 = wbf + (size_t)(m0 + rB + 64) * CINL + kB;

    // LDS byte pointers; write side leads read side by one buffer
    char*       aWr = (char*)&Asub[0][rB][0] + kB * 2;
    char*       bWr = (char*)&Bsub[0][rB][0] + kB * 2;
    const char* aRd = (const char*)&Asub[0][wm * 32 + lm][0] + half * 16;
    const char* bRd = (const char*)&Bsub[0][wn * 32 + lm][0] + half * 16;
    int dAw = 8192, dBw = 4096;          // toggle deltas (sign flips each use)
    int dAr = 8192, dBr = 4096;

    // stage coordinates for the NEXT chunk to stage (advanced incrementally)
    size_t aoff = 0;
    int s_kh = 0, s_kw = 0, s_c0 = 0;

    auto stage = [&]() {
        // A: weights [kpos][cout][cin] -> Asub[m][k] (running offset, no mul)
        *(v8bf*)aWr              = *(const v8bf*)(aptr0 + aoff);
        *(v8bf*)(aWr + 64 * 64)  = *(const v8bf*)(aptr1 + aoff);
        // B: activations NHWC with im2col shift -> Bsub[n][k]
        const int py = py0 + s_kh - PAD;
        const int px = rB + s_kw - PAD;
        const bool ok = ((unsigned)py < (unsigned)H_) && ((unsigned)px < (unsigned)W_);
        const int pyc = clampi(py, 0, H_ - 1);
        const int pxc = clampi(px, 0, W_ - 1);
        v8bf bv = *(const v8bf*)(xT + ((size_t)(pyc * W_ + pxc)) * CINL + s_c0 + kB);
        bv = ok ? bv : (v8bf){};
        *(v8bf*)bWr = bv;
        // toggle write buffers
        aWr += dAw; bWr += dBw; dAw = -dAw; dBw = -dBw;
        // advance chunk coords (constant deltas only)
        s_c0 += 32;
        if (s_c0 == CINL) {
            s_c0 = 0;
            aoff += A_WRAP;
            ++s_kw;
            if (s_kw == KSZ) { s_kw = 0; ++s_kh; }
        } else {
            aoff += A_STEP;
        }
        __builtin_prefetch(aptr0 + aoff, 0, 1);   // next chunk's weights
    };

    auto fragA = [&](const char* p) -> v16bf {    // p includes half*16
        v8bf lo = *(const v8bf*)(p);
        v8bf hi = *(const v8bf*)(p + 32);
        return __builtin_shufflevector(lo, hi,
                    0,1,2,3,4,5,6,7,8,9,10,11,12,13,14,15);
    };
    auto fragB = [&](const char* p) -> v16bf {    // p includes half*16
        v8bf lo = *(const v8bf*)(p);
        v8bf hi = *(const v8bf*)(p + 32);
        return __builtin_shufflevector(lo, hi,
                    0,1,2,3,4,5,6,7,8,9,10,11,12,13,14,15);
    };

    acc[0][0] = (v8f){}; acc[0][1] = (v8f){};
    acc[1][0] = (v8f){}; acc[1][1] = (v8f){};

    stage();
#pragma unroll 1
    for (int ch = 0; ch < NCH; ++ch) {
        __syncthreads();
        if (ch + 1 < NCH) stage();

        // issue all fragment loads, then all WMMAs
        v16bf a0 = fragA(aRd);
        v16bf a1 = fragA(aRd + 16 * 64);     // +16 rows
        v16bf b0 = fragB(bRd);
        v16bf b1 = fragB(bRd + 16 * 64);     // +16 rows
        aRd += dAr; bRd += dBr; dAr = -dAr; dBr = -dBr;

        acc[0][0] = __builtin_amdgcn_wmma_f32_16x16x32_bf16(
                        false, a0, false, b0, (short)0, acc[0][0], false, false);
        acc[0][1] = __builtin_amdgcn_wmma_f32_16x16x32_bf16(
                        false, a0, false, b1, (short)0, acc[0][1], false, false);
        acc[1][0] = __builtin_amdgcn_wmma_f32_16x16x32_bf16(
                        false, a1, false, b0, (short)0, acc[1][0], false, false);
        acc[1][1] = __builtin_amdgcn_wmma_f32_16x16x32_bf16(
                        false, a1, false, b1, (short)0, acc[1][1], false, false);
    }
}

// ---------------- conv1: x2h (5x5, pad 2) -> raw y, NHWC ----------------
__global__ void __launch_bounds__(256)
conv_x2h_kernel(const __bf16* __restrict__ xT, const __bf16* __restrict__ wbf,
                const float* __restrict__ bias, float* __restrict__ y_raw) {
    const int py0 = blockIdx.x;                 // one image row per block
    const int m0  = blockIdx.y * 128;
    const int tn  = blockIdx.z;

    v8f acc[2][2];
    conv_tile_acc<CIN, 5, 2>(wbf, C3, xT + (size_t)tn * CIN * HW, m0, py0, acc);

    const int lane = threadIdx.x & 31, wave = threadIdx.x >> 5;
    const int wm = wave & 3, wn = wave >> 2, half = lane >> 4, lm = lane & 15;
#pragma unroll
    for (int ma = 0; ma < 2; ++ma) {
        const int cb = m0 + wm * 32 + ma * 16 + 8 * half;
        v4f bl = *(const v4f*)&bias[cb];
        v4f bh = *(const v4f*)&bias[cb + 4];
#pragma unroll
        for (int nb = 0; nb < 2; ++nb) {
            v8f a = acc[ma][nb];
            int pix = py0 * W_ + wn * 32 + nb * 16 + lm;
            float* dst = y_raw + ((size_t)tn * HW + pix) * C3 + cb;
            v4f lo, hi;
#pragma unroll
            for (int r = 0; r < 4; ++r) { lo[r] = a[r] + bl[r]; hi[r] = a[r + 4] + bh[r]; }
            *(v4f*)dst = lo;
            *(v4f*)(dst + 4) = hi;
        }
    }
}

// ---------------- BN stats (coalesced column reduce + atomics) ----------------
__global__ void zero_stats_kernel(float* __restrict__ sums) {
    int i = blockIdx.x * 256 + threadIdx.x;
    if (i < 2 * C3) sums[i] = 0.0f;
}

__global__ void __launch_bounds__(C3)
bn_partial_kernel(const float* __restrict__ y_raw, float* __restrict__ sums) {
    const int c = threadIdx.x;                  // 384 threads: one channel each
    float s = 0.f, ss = 0.f;
    const int rows = T_ * NB * HW;
    for (int pix = blockIdx.x; pix < rows; pix += gridDim.x) {
        float v = y_raw[(size_t)pix * C3 + c];  // consecutive lanes -> consecutive addrs
        s += v; ss += v * v;
    }
    atomicAdd(&sums[c], s);
    atomicAdd(&sums[C3 + c], ss);
}

__global__ void __launch_bounds__(C3)
bn_finalize_kernel(const float* __restrict__ sums, const float* __restrict__ gamma,
                   const float* __restrict__ beta, float* __restrict__ scale,
                   float* __restrict__ shift) {
    const int c = threadIdx.x;
    const float inv_n = 1.0f / (float)(T_ * NB * HW);
    float mean = sums[c] * inv_n;
    float var  = sums[C3 + c] * inv_n - mean * mean;
    float sc   = gamma[c] * rsqrtf(var + BN_EPS);
    scale[c] = sc;
    shift[c] = beta[c] - mean * sc;
}

// ---------------- h0 = sigmoid(y0[:, :C]) * tanh(y0[:, 2C:]) ----------------
__global__ void __launch_bounds__(256)
h0_kernel(const float* __restrict__ y_raw, const float* __restrict__ scale,
          const float* __restrict__ shift, float* __restrict__ out,
          float* __restrict__ h_f32, __bf16* __restrict__ h_bf) {
    int g = blockIdx.x * 256 + threadIdx.x;     // NB*HW*C/8 groups of 8 channels
    int cb = (g & (C / 8 - 1)) * 8;
    int rest = g >> 4;
    int pix = rest & (HW - 1);
    int n = rest >> 12;
    const float* yb = y_raw + ((size_t)n * HW + pix) * C3;  // t=0 -> tn=n
    float h8[8];
    v8bf hb;
#pragma unroll
    for (int j = 0; j < 8; ++j) {
        int cz = cb + j, chn = 2 * C + cb + j;
        float yz = yb[cz] * scale[cz] + shift[cz];
        float yh = yb[chn] * scale[chn] + shift[chn];
        float h = sigmoidf_(yz) * tanhf(yh);
        h8[j] = h;
        hb[j] = f2bf(h);
        out[((size_t)n * C + cz) * HW + pix] = h;           // NCHW output
    }
    float* hf = h_f32 + ((size_t)n * HW + pix) * C + cb;
    *(v4f*)hf = (v4f){h8[0], h8[1], h8[2], h8[3]};
    *(v4f*)(hf + 4) = (v4f){h8[4], h8[5], h8[6], h8[7]};
    *(v8bf*)(h_bf + ((size_t)n * HW + pix) * C + cb) = hb;
}

// ---------------- h2zr conv (3x3, pad 1): z, r*h ----------------
__global__ void __launch_bounds__(256)
conv_h2zr_kernel(const __bf16* __restrict__ h_bf, const __bf16* __restrict__ wbf,
                 const float* __restrict__ bias, const float* __restrict__ y_raw,
                 const float* __restrict__ scale, const float* __restrict__ shift,
                 const float* __restrict__ h_f32, float* __restrict__ z_buf,
                 __bf16* __restrict__ rh_bf, int t) {
    const int py0 = blockIdx.x;
    const int m0  = blockIdx.y * 128;           // 0 -> z half, 128 -> r half
    const int n   = blockIdx.z;

    v8f acc[2][2];
    conv_tile_acc<C, 3, 1>(wbf, C2, h_bf + (size_t)n * C * HW, m0, py0, acc);

    const int lane = threadIdx.x & 31, wave = threadIdx.x >> 5;
    const int wm = wave & 3, wn = wave >> 2, half = lane >> 4, lm = lane & 15;
    const int tn = t * NB + n;
#pragma unroll
    for (int ma = 0; ma < 2; ++ma) {
        const int cb = m0 + wm * 32 + ma * 16 + 8 * half;
#pragma unroll
        for (int nb = 0; nb < 2; ++nb) {
            v8f a = acc[ma][nb];
            int pix = py0 * W_ + wn * 32 + nb * 16 + lm;
            const float* yb = y_raw + ((size_t)tn * HW + pix) * C3 + cb;
            float sg[8];
#pragma unroll
            for (int j = 0; j < 8; ++j) {
                int c = cb + j;
                float xn = yb[j] * scale[c] + shift[c];
                sg[j] = sigmoidf_(a[j] + bias[c] + xn);
            }
            if (m0 == 0) {   // z half
                float* zd = z_buf + ((size_t)n * HW + pix) * C + cb;
                *(v4f*)zd = (v4f){sg[0], sg[1], sg[2], sg[3]};
                *(v4f*)(zd + 4) = (v4f){sg[4], sg[5], sg[6], sg[7]};
            } else {         // r half -> r*h (bf16, NHWC)
                int c0 = cb - C;
                const float* hp = h_f32 + ((size_t)n * HW + pix) * C + c0;
                v8bf rh;
#pragma unroll
                for (int j = 0; j < 8; ++j) rh[j] = f2bf(sg[j] * hp[j]);
                *(v8bf*)(rh_bf + ((size_t)n * HW + pix) * C + c0) = rh;
            }
        }
    }
}

// ---------------- h2h conv (3x3, pad 1): GRU update ----------------
__global__ void __launch_bounds__(256)
conv_h2h_kernel(const __bf16* __restrict__ rh_bf, const __bf16* __restrict__ wbf,
                const float* __restrict__ bias, const float* __restrict__ y_raw,
                const float* __restrict__ scale, const float* __restrict__ shift,
                const float* __restrict__ z_buf, float* __restrict__ out,
                float* __restrict__ h_f32, __bf16* __restrict__ h_bf, int t) {
    const int py0 = blockIdx.x;
    const int n   = blockIdx.z;

    v8f acc[2][2];
    conv_tile_acc<C, 3, 1>(wbf, C, rh_bf + (size_t)n * C * HW, 0, py0, acc);

    const int lane = threadIdx.x & 31, wave = threadIdx.x >> 5;
    const int wm = wave & 3, wn = wave >> 2, half = lane >> 4, lm = lane & 15;
    const int tn = t * NB + n;
#pragma unroll
    for (int ma = 0; ma < 2; ++ma) {
        const int cb = wm * 32 + ma * 16 + 8 * half;
#pragma unroll
        for (int nb = 0; nb < 2; ++nb) {
            v8f a = acc[ma][nb];
            int pix = py0 * W_ + wn * 32 + nb * 16 + lm;
            const float* yb = y_raw + ((size_t)tn * HW + pix) * C3 + 2 * C + cb;
            const float* zp = z_buf + ((size_t)n * HW + pix) * C + cb;
            float* hf = h_f32 + ((size_t)n * HW + pix) * C + cb;
            float hn8[8];
            v8bf hb;
#pragma unroll
            for (int j = 0; j < 8; ++j) {
                int c = cb + j, chn = 2 * C + c;
                float xn = yb[j] * scale[chn] + shift[chn];
                float ht = tanhf(a[j] + bias[c] + xn);
                float z  = zp[j];
                float hn = (1.0f - z) * hf[j] + z * ht;
                hn8[j] = hn;
                hb[j] = f2bf(hn);
                out[(((size_t)t * NB + n) * C + c) * HW + pix] = hn;   // NCHW output
            }
            *(v4f*)hf = (v4f){hn8[0], hn8[1], hn8[2], hn8[3]};
            *(v4f*)(hf + 4) = (v4f){hn8[4], hn8[5], hn8[6], hn8[7]};
            *(v8bf*)(h_bf + ((size_t)n * HW + pix) * C + cb) = hb;
        }
    }
}

// ---------------- launcher ----------------
extern "C" void kernel_launch(void* const* d_in, const int* in_sizes, int n_in,
                              void* d_out, int out_size, void* d_ws, size_t ws_size,
                              hipStream_t stream) {
    const float* x      = (const float*)d_in[0];
    const float* w_x2h  = (const float*)d_in[1];
    const float* b_x2h  = (const float*)d_in[2];
    const float* gamma  = (const float*)d_in[3];
    const float* beta   = (const float*)d_in[4];
    const float* w_h2zr = (const float*)d_in[5];
    const float* b_h2zr = (const float*)d_in[6];
    const float* w_h2h  = (const float*)d_in[7];
    const float* b_h2h  = (const float*)d_in[8];
    float* out = (float*)d_out;

    char* base = (char*)d_ws;
    size_t off = 0;
    auto carve = [&](size_t bytes) -> char* {
        char* p = base + off;
        off = (off + bytes + 255) & ~(size_t)255;
        return p;
    };
    float*  y_raw  = (float*)carve((size_t)T_ * NB * HW * C3 * sizeof(float));  // NHWC
    float*  sums   = (float*)carve(2 * C3 * sizeof(float));
    float*  scale  = (float*)carve(C3 * sizeof(float));
    float*  shift  = (float*)carve(C3 * sizeof(float));
    float*  h_f32  = (float*)carve((size_t)NB * HW * C * sizeof(float));        // NHWC
    float*  z_buf  = (float*)carve((size_t)NB * HW * C * sizeof(float));        // NHWC
    __bf16* h_bf   = (__bf16*)carve((size_t)NB * HW * C * sizeof(__bf16));      // NHWC
    __bf16* rh_bf  = (__bf16*)carve((size_t)NB * HW * C * sizeof(__bf16));      // NHWC
    __bf16* x_bf   = (__bf16*)carve((size_t)T_ * NB * HW * CIN * sizeof(__bf16)); // NHWC
    __bf16* wx_bf  = (__bf16*)carve((size_t)25 * C3 * CIN * sizeof(__bf16));
    __bf16* wzr_bf = (__bf16*)carve((size_t)9 * C2 * C * sizeof(__bf16));
    __bf16* wh_bf  = (__bf16*)carve((size_t)9 * C * C * sizeof(__bf16));
    (void)ws_size; (void)in_sizes; (void)n_in; (void)out_size;

    // 0) layout conversions (f32 -> bf16, k-major / NHWC)
    {
        int nx = T_ * NB * CIN * HW;
        cvt_x_nhwc<<<(nx + 255) / 256, 256, 0, stream>>>(x, x_bf, nx);
        int nw1 = C3 * CIN * 25;
        cvt_w_kmaj<<<(nw1 + 255) / 256, 256, 0, stream>>>(w_x2h, wx_bf, C3, CIN, 5);
        int nw2 = C2 * C * 9;
        cvt_w_kmaj<<<(nw2 + 255) / 256, 256, 0, stream>>>(w_h2zr, wzr_bf, C2, C, 3);
        int nw3 = C * C * 9;
        cvt_w_kmaj<<<(nw3 + 255) / 256, 256, 0, stream>>>(w_h2h, wh_bf, C, C, 3);
    }

    // 1) front conv: implicit GEMM, WMMA bf16
    conv_x2h_kernel<<<dim3(H_, C3 / 128, T_ * NB), 256, 0, stream>>>(
        x_bf, wx_bf, b_x2h, y_raw);

    // 2) BN stats -> per-channel scale/shift
    zero_stats_kernel<<<(2 * C3 + 255) / 256, 256, 0, stream>>>(sums);
    bn_partial_kernel<<<1024, C3, 0, stream>>>(y_raw, sums);
    bn_finalize_kernel<<<1, C3, 0, stream>>>(sums, gamma, beta, scale, shift);

    // 3) h0
    h0_kernel<<<(NB * HW * C / 8) / 256, 256, 0, stream>>>(
        y_raw, scale, shift, out, h_f32, h_bf);

    // 4) recurrence (working set is L2-resident)
    for (int t = 1; t < T_; ++t) {
        conv_h2zr_kernel<<<dim3(H_, C2 / 128, NB), 256, 0, stream>>>(
            h_bf, wzr_bf, b_h2zr, y_raw, scale, shift, h_f32, z_buf, rh_bf, t);
        conv_h2h_kernel<<<dim3(H_, 1, NB), 256, 0, stream>>>(
            rh_bf, wh_bf, b_h2h, y_raw, scale, shift, z_buf, out, h_f32, h_bf, t);
    }
}
